// LMMDrift_55456617726509
// MI455X (gfx1250) — matrix operand order repaired and checked by reading, not compiled, not run on hardware
//
#include <hip/hip_runtime.h>
#include <hip/hip_bf16.h>

// CDNA5 (gfx1250, wave32) implementation of the LMM drift.
//
// Key factorization: drift_sum[i] = L[i] . ( sum_{j=eta..i} g'[j] L[j] )
// with g'[j] = alive[j]*g[j], alive[j] = (t <= tenor[j]).
// => O(N*K) via a K-vector prefix scan over 16-row blocks, plus an
//    intra-block triangular 16x16 product done with v_wmma_f32_16x16x32_f16.

typedef __attribute__((ext_vector_type(16))) _Float16 v16h;
typedef __attribute__((ext_vector_type(8)))  float    v8f;

#define TILE 16

// ---------------------------------------------------------------------------
// Kernel A: per-forward precompute of sigma and g' (= alive * g)
// ---------------------------------------------------------------------------
__global__ __launch_bounds__(256)
void lmm_precompute(const float* __restrict__ t_,
                    const float* __restrict__ y,
                    const float* __restrict__ tau_,
                    const float* __restrict__ tenor,
                    const float* __restrict__ va, const float* __restrict__ vb,
                    const float* __restrict__ vc, const float* __restrict__ vd,
                    float* __restrict__ sigma_out,
                    float* __restrict__ gp_out,
                    int N)
{
    int i = blockIdx.x * blockDim.x + threadIdx.x;
    if (i >= N) return;
    float t = t_[0];
    float a = va[0], b = vb[0], c = vc[0], d = vd[0];
    float Ti  = tenor[i];
    float ttm = fmaxf(Ti - t, 0.0f);
    float sig = (a + b * ttm) * __expf(-c * ttm) + d;      // Rebonato abcd
    float F   = __expf(y[i]);
    float tau = tau_[i];
    float g   = tau * sig * F / (1.0f + tau * F);
    float alive = (t <= Ti) ? 1.0f : 0.0f;
    sigma_out[i] = sig;
    gp_out[i]    = alive * g;                              // mask j>=eta folded in
}

// ---------------------------------------------------------------------------
// Kernel B1: per-16-row-block partial K-vectors  P[B][k] = sum_{r} g'[j] L[j][k]
// grid = NB blocks, 64 threads (one per k)
// ---------------------------------------------------------------------------
__global__ __launch_bounds__(64)
void lmm_block_partial(const float* __restrict__ L,
                       const float* __restrict__ gp,
                       float* __restrict__ P,
                       int K)
{
    int B  = blockIdx.x;
    int k  = threadIdx.x;
    int i0 = B * TILE;
    float s = 0.0f;
#pragma unroll
    for (int r = 0; r < TILE; ++r)
        s += gp[i0 + r] * L[(size_t)(i0 + r) * K + k];     // coalesced across k
    P[(size_t)B * K + k] = s;
}

// ---------------------------------------------------------------------------
// Kernel B2: exclusive prefix scan over blocks of K-vectors (single workgroup,
// thread k scans lane k; 512 coalesced 64-thread loads)
// ---------------------------------------------------------------------------
__global__ __launch_bounds__(64)
void lmm_block_scan(const float* __restrict__ P,
                    float* __restrict__ Spre,
                    int NB, int K)
{
    int k = threadIdx.x;
    float run = 0.0f;
#pragma unroll 8
    for (int B = 0; B < NB; ++B) {
        float v = P[(size_t)B * K + k];
        Spre[(size_t)B * K + k] = run;
        run += v;
    }
}

// ---------------------------------------------------------------------------
// Kernel C: one wave (32 threads) per 16-row block.
//   intra-block triangular term via 2x v_wmma_f32_16x16x32_f16 (K=64 total)
//   inter-block term: L[i] . Spre[B]  (f32 dot, split across half-waves)
// ---------------------------------------------------------------------------
__global__ __launch_bounds__(32)
void lmm_drift_main(const float* __restrict__ L,
                    const float* __restrict__ gp,
                    const float* __restrict__ sigma,
                    const float* __restrict__ Spre,
                    const float* __restrict__ tenor,
                    const float* __restrict__ t_,
                    float* __restrict__ out,
                    int K)
{
    __shared__ float lds_m[TILE * (TILE + 1)];   // masked 16x16 intra-block matrix
    __shared__ float red_dot[32];
    __shared__ float red_vs[32];

    const int B  = blockIdx.x;
    const int l  = threadIdx.x;        // 0..31, full wave, no divergence pre-WMMA
    const int m  = l & 15;             // row (A) / col (B) index within tile
    const int hi = l >> 4;             // half-wave selector
    const int i0 = B * TILE;

    const float* rowL = L + (size_t)(i0 + m) * K;    // row for A fragment
    __builtin_prefetch(Spre + (size_t)B * K, 0, 0);  // -> global_prefetch_b8

    // --- A fragments: 16-bit A-matrix 16x32 layout (ISA 7.12.2) ---
    // lane m (lo half): elems 0..7 -> K=0..7, elems 8..15 -> K=16..23
    // lane m (hi half): elems 0..7 -> K=8..15, elems 8..15 -> K=24..31
    v16h a0, a1;
#pragma unroll
    for (int e = 0; e < 16; ++e) {
        int kk = (e < 8) ? (hi * 8 + e) : (16 + hi * 8 + (e - 8));
        a0[e] = (_Float16)rowL[kk];
        a1[e] = (_Float16)rowL[32 + kk];
    }

    // --- B fragments: B = (g' ⊙ L)^T, 32x16 per fragment ---
    // lane n holds column n; elems 0..7 -> K=8*hi+0..7, elems 8..15 -> K=16+8*hi+0..7
    const float* colL = L + (size_t)(i0 + m) * K;    // n == m lane mapping
    const float  gcol = gp[i0 + m];                  // alive mask already folded
    v16h b0, b1;
#pragma unroll
    for (int e = 0; e < 16; ++e) {
        int kk = (e < 8) ? (hi * 8 + e) : (16 + hi * 8 + (e - 8));
        b0[e] = (_Float16)(gcol * colL[kk]);
        b1[e] = (_Float16)(gcol * colL[32 + kk]);
    }

    // --- M = A x B accumulated over K=64 in f32 ---
    v8f acc = {};
    acc = __builtin_amdgcn_wmma_f32_16x16x32_f16(false, a0, false, b0,
                                                 (short)0, acc, false, false);
    acc = __builtin_amdgcn_wmma_f32_16x16x32_f16(false, a1, false, b1,
                                                 (short)0, acc, false, false);

    // --- masked (col <= row) scatter of D fragment to LDS ---
    // D layout: VGPR v, lanes 0-15: row=v,   col=lane
    //                    lanes 16-31: row=v+8, col=lane-16
#pragma unroll
    for (int v = 0; v < 8; ++v) {
        int row = v + 8 * hi;
        int col = m;
        lds_m[row * (TILE + 1) + col] = (col <= row) ? acc[v] : 0.0f;
    }

    // --- inter-block dot L[i].Spre[B] and vol_sq = |L[i]|^2, half-wave split ---
    const float* sp = Spre + (size_t)B * K;
    float dotp = 0.0f, vsp = 0.0f;
#pragma unroll 8
    for (int q = 0; q < 32; ++q) {
        int kk   = hi * 32 + q;
        float lv = rowL[kk];
        dotp += lv * sp[kk];
        vsp  += lv * lv;
    }
    red_dot[l] = dotp;
    red_vs[l]  = vsp;
    __syncthreads();

    if (l < TILE) {
        float rowsum = 0.0f;
#pragma unroll
        for (int c = 0; c < TILE; ++c)
            rowsum += lds_m[l * (TILE + 1) + c];

        float dot  = red_dot[l] + red_dot[l + 16];
        float vol2 = red_vs[l]  + red_vs[l + 16];

        int   i     = i0 + l;
        float sig   = sigma[i];
        float alive = (t_[0] <= tenor[i]) ? 1.0f : 0.0f;
        float drift = dot + rowsum;                 // = sum_{eta<=j<=i} corr*g
        out[i] = alive * (sig * drift - 0.5f * sig * sig * vol2);
    }
}

// ---------------------------------------------------------------------------
extern "C" void kernel_launch(void* const* d_in, const int* in_sizes, int n_in,
                              void* d_out, int out_size, void* d_ws, size_t ws_size,
                              hipStream_t stream)
{
    const float* t_    = (const float*)d_in[0];
    const float* y     = (const float*)d_in[1];
    const float* tau_  = (const float*)d_in[2];
    const float* L     = (const float*)d_in[3];
    const float* tenor = (const float*)d_in[4];
    const float* va    = (const float*)d_in[5];
    const float* vb    = (const float*)d_in[6];
    const float* vc    = (const float*)d_in[7];
    const float* vd    = (const float*)d_in[8];

    const int N  = in_sizes[1];
    const int K  = in_sizes[3] / N;     // 64
    const int NB = N / TILE;            // 512

    float* ws    = (float*)d_ws;
    float* sigma = ws;                  // N
    float* gp    = sigma + N;           // N
    float* P     = gp + N;              // NB*K
    float* Spre  = P + (size_t)NB * K;  // NB*K

    lmm_precompute<<<(N + 255) / 256, 256, 0, stream>>>(
        t_, y, tau_, tenor, va, vb, vc, vd, sigma, gp, N);

    lmm_block_partial<<<NB, K, 0, stream>>>(L, gp, P, K);

    lmm_block_scan<<<1, K, 0, stream>>>(P, Spre, NB, K);

    lmm_drift_main<<<NB, 32, 0, stream>>>(
        L, gp, sigma, Spre, tenor, t_, (float*)d_out, K);
}
